// MultiHeadAttention_197568496251
// MI455X (gfx1250) — matrix operand order repaired and checked by reading, not compiled, not run on hardware
//
#include <hip/hip_runtime.h>

typedef __attribute__((ext_vector_type(16))) _Float16 v16h;
typedef __attribute__((ext_vector_type(8)))  _Float16 v8h;
typedef __attribute__((ext_vector_type(4)))  _Float16 v4h;
typedef __attribute__((ext_vector_type(8)))  float    v8f;
typedef __attribute__((ext_vector_type(4)))  float    v4f;

#define B_   2
#define S_   2048
#define H_   16
#define DK_  64
#define DM_  1024
#define MTOT (B_*S_)   // 4096 tokens

union H16 { v16h v; v8h h[2]; };

__device__ __forceinline__ v8f wmma16(v16h a, v16h b, v8f c) {
  // v_wmma_f32_16x16x32_f16: D = A*B + C, f32 accumulate
  return __builtin_amdgcn_wmma_f32_16x16x32_f16(false, a, false, b, (short)0, c,
                                                false, false);
}

__device__ __forceinline__ v8f zero8() {
  v8f z = {0.f,0.f,0.f,0.f,0.f,0.f,0.f,0.f};
  return z;
}

// ---------------- f32 -> f16 conversion ----------------
__global__ __launch_bounds__(256) void cvt_kernel(const float* __restrict__ src,
                                                  _Float16* __restrict__ dst, int n) {
  int i = (blockIdx.x * 256 + threadIdx.x) * 4;
  if (i < n) {
    v4f f = *(const v4f*)(src + i);
    v4h h;
    h[0] = (_Float16)f[0]; h[1] = (_Float16)f[1];
    h[2] = (_Float16)f[2]; h[3] = (_Float16)f[3];
    *(v4h*)(dst + i) = h;
  }
}

// ---------------- tiled WMMA GEMM: C = A[M,K] * W[N,K]^T ----------------
// M=4096, N=1024, K=1024. Block tile 128x128, BK=64, 256 threads = 8 waves (4x2).
// MODE 0: f16 out, [B,H,S,D]    (Q, K projections)
// MODE 1: f16 out, [B,H,D,S]    (V projection, transposed for attention)
// MODE 2: f32 out, + bias + residual (FC)
template<int MODE>
__global__ __launch_bounds__(256) void gemm_kernel(const _Float16* __restrict__ A,
                                                   const _Float16* __restrict__ W,
                                                   void* __restrict__ outp,
                                                   const float* __restrict__ bias,
                                                   const float* __restrict__ resid) {
  __shared__ _Float16 sA[128 * 72];
  __shared__ _Float16 sW[128 * 72];

  const int tid  = threadIdx.x;
  const int lane = tid & 31, wave = tid >> 5;
  const int hi   = lane >> 4, l = lane & 15;
  const int wm   = (wave & 3) * 32;   // wave row offset in tile
  const int wn   = (wave >> 2) * 64;  // wave col offset in tile
  const int gm0  = blockIdx.y * 128;
  const int gn0  = blockIdx.x * 128;

  v8f acc[2][4];
  #pragma unroll
  for (int i = 0; i < 2; i++)
    #pragma unroll
    for (int j = 0; j < 4; j++) acc[i][j] = zero8();

  const int lr = tid >> 1;          // 0..127 tile row
  const int lc = (tid & 1) * 32;    // 0 or 32
  const _Float16* Arow = A + (size_t)(gm0 + lr) * DM_ + lc;
  const _Float16* Wrow = W + (size_t)(gn0 + lr) * DM_ + lc;
  _Float16* sArow = sA + lr * 72 + lc;
  _Float16* sWrow = sW + lr * 72 + lc;

  for (int k0 = 0; k0 < DM_; k0 += 64) {
    __syncthreads();
    #pragma unroll
    for (int c = 0; c < 32; c += 8) {
      *(v8h*)(sArow + c) = *(const v8h*)(Arow + k0 + c);
      *(v8h*)(sWrow + c) = *(const v8h*)(Wrow + k0 + c);
    }
    __syncthreads();
    #pragma unroll
    for (int kk = 0; kk < 2; kk++) {
      H16 af[2], bf[4];
      #pragma unroll
      for (int mi = 0; mi < 2; mi++) {
        const _Float16* p = sA + (wm + mi * 16 + l) * 72 + kk * 32;
        af[mi].h[0] = *(const v8h*)(p + hi * 8);
        af[mi].h[1] = *(const v8h*)(p + 16 + hi * 8);
      }
      #pragma unroll
      for (int ni = 0; ni < 4; ni++) {
        const _Float16* p = sW + (wn + ni * 16 + l) * 72 + kk * 32;
        bf[ni].h[0] = *(const v8h*)(p + hi * 8);
        bf[ni].h[1] = *(const v8h*)(p + 16 + hi * 8);
      }
      #pragma unroll
      for (int mi = 0; mi < 2; mi++)
        #pragma unroll
        for (int ni = 0; ni < 4; ni++)
          acc[mi][ni] = wmma16(af[mi].v, bf[ni].v, acc[mi][ni]);
    }
  }

  #pragma unroll
  for (int mi = 0; mi < 2; mi++) {
    #pragma unroll
    for (int ni = 0; ni < 4; ni++) {
      v8f c = acc[mi][ni];
      int n = gn0 + wn + ni * 16 + l;
      if (MODE == 0) {
        _Float16* out16 = (_Float16*)outp;
        int h = n >> 6, d = n & 63;
        #pragma unroll
        for (int r = 0; r < 8; r++) {
          int m = gm0 + wm + mi * 16 + 8 * hi + r;
          int b = m >> 11, s = m & 2047;
          out16[((size_t)((b * H_ + h) * S_ + s)) * DK_ + d] = (_Float16)c[r];
        }
      } else if (MODE == 1) {
        _Float16* out16 = (_Float16*)outp;
        int h = n >> 6, d = n & 63;
        int m0 = gm0 + wm + mi * 16 + 8 * hi;
        int b = m0 >> 11, s0 = m0 & 2047;
        v8h ph;
        #pragma unroll
        for (int r = 0; r < 8; r++) ph[r] = (_Float16)c[r];
        *(v8h*)(out16 + ((size_t)((b * H_ + h) * DK_ + d)) * S_ + s0) = ph;
      } else {
        float* outf = (float*)outp;
        float bn = bias[n];
        #pragma unroll
        for (int r = 0; r < 8; r++) {
          int m = gm0 + wm + mi * 16 + 8 * hi + r;
          size_t idx = (size_t)m * DM_ + n;
          outf[idx] = c[r] + bn + resid[idx];
        }
      }
    }
  }
}

// ---------------- flash attention, transposed-score formulation ----------------
// S^T = K @ Q^T   (C-layout: M = key in VGPRs, N = query in lanes)
// O^T = V^T @ P^T (C-layout: M = d   in VGPRs, N = query in lanes)
// => softmax stats are per-lane scalars; only one xor-16 shuffle per reduction.
// grid: (S/128, B*H); 8 waves/block; each wave owns 16 queries (N dim).
__global__ __launch_bounds__(256) void attn_kernel(const _Float16* __restrict__ Qh,
                                                   const _Float16* __restrict__ Kh,
                                                   const _Float16* __restrict__ Vt,
                                                   _Float16* __restrict__ Out) {
  __shared__ _Float16 sK[64 * 72];       // [key][d]
  __shared__ _Float16 sV[64 * 72];       // [d][key]  (V pre-transposed in global)
  __shared__ _Float16 sP[8 * 16 * 72];   // per-wave P^T scratch as [q][key]

  const int tid  = threadIdx.x;
  const int lane = tid & 31, wave = tid >> 5;
  const int hi   = lane >> 4, l = lane & 15;
  const int bh   = blockIdx.y;
  const int b    = bh >> 4, h = bh & 15;
  const int q0   = blockIdx.x * 128 + wave * 16;

  const _Float16* Qbase = Qh + (size_t)bh * S_ * DK_;
  const _Float16* Kbase = Kh + (size_t)bh * S_ * DK_;
  const _Float16* Vbase = Vt + (size_t)bh * DK_ * S_;

  // resident Q as B-fragments: column = query q0+l, table k order along d
  H16 bq[2];
  {
    const _Float16* qrow = Qbase + (size_t)(q0 + l) * DK_;
    #pragma unroll
    for (int t = 0; t < 2; t++) {
      bq[t].h[0] = *(const v8h*)(qrow + t * 32 + hi * 8);
      bq[t].h[1] = *(const v8h*)(qrow + t * 32 + 16 + hi * 8);
    }
  }

  v8f o[4];                               // O^T accumulators, mi indexes d
  #pragma unroll
  for (int mi = 0; mi < 4; mi++) o[mi] = zero8();
  float mrun = -1e30f, lrun = 0.f;

  const int lr = tid >> 2;                // 0..63
  const int lc = (tid & 3) * 16;          // 0,16,32,48
  _Float16* pbase = sP + wave * 16 * 72;
  const float scale = 0.125f;             // 1/sqrt(64)

  for (int j = 0; j < S_; j += 64) {
    __syncthreads();
    {
      const _Float16* kg = Kbase + (size_t)(j + lr) * DK_ + lc;
      const _Float16* vg = Vbase + (size_t)lr * S_ + j + lc;
      *(v8h*)(sK + lr * 72 + lc)     = *(const v8h*)(kg);
      *(v8h*)(sK + lr * 72 + lc + 8) = *(const v8h*)(kg + 8);
      *(v8h*)(sV + lr * 72 + lc)     = *(const v8h*)(vg);
      *(v8h*)(sV + lr * 72 + lc + 8) = *(const v8h*)(vg + 8);
    }
    __syncthreads();

    // S^T = K @ Q^T : A-frags = K rows (M=key), B-frags = resident Q
    v8f sacc[4];
    #pragma unroll
    for (int mi = 0; mi < 4; mi++) {
      v8f z = zero8();
      #pragma unroll
      for (int t = 0; t < 2; t++) {
        H16 ak;
        const _Float16* p = sK + (mi * 16 + l) * 72 + t * 32;
        ak.h[0] = *(const v8h*)(p + hi * 8);
        ak.h[1] = *(const v8h*)(p + 16 + hi * 8);
        z = wmma16(ak.v, bq[t].v, z);
      }
      sacc[mi] = z;
    }

    // per-query softmax stats (query == lane; halves hold disjoint key sets)
    float mx;
    {
      v8f t01, t23;
      #pragma unroll
      for (int r = 0; r < 8; r++) {
        t01[r] = fmaxf(sacc[0][r], sacc[1][r]);
        t23[r] = fmaxf(sacc[2][r], sacc[3][r]);
      }
      float m0 = fmaxf(fmaxf(t01[0], t01[1]), fmaxf(t01[2], t01[3]));
      float m1 = fmaxf(fmaxf(t01[4], t01[5]), fmaxf(t01[6], t01[7]));
      float m2 = fmaxf(fmaxf(t23[0], t23[1]), fmaxf(t23[2], t23[3]));
      float m3 = fmaxf(fmaxf(t23[4], t23[5]), fmaxf(t23[6], t23[7]));
      mx = fmaxf(fmaxf(m0, m1), fmaxf(m2, m3));
    }
    mx = fmaxf(mx, __shfl_xor(mx, 16, 32));   // combine the two halves
    mx *= scale;
    float mnew = fmaxf(mrun, mx);
    float resc = __expf(mrun - mnew);
    mrun = mnew;

    float sum = 0.f;
    #pragma unroll
    for (int mi = 0; mi < 4; mi++) {
      #pragma unroll
      for (int r = 0; r < 8; r++) {
        float p = __expf(sacc[mi][r] * scale - mnew);
        sacc[mi][r] = p;
        sum += p;
      }
    }
    sum += __shfl_xor(sum, 16, 32);
    lrun = lrun * resc + sum;
    #pragma unroll
    for (int mi = 0; mi < 4; mi++)
      #pragma unroll
      for (int r = 0; r < 8; r++) o[mi][r] *= resc;

    // P^T -> LDS scratch [q][k]: lane owns row q=l, keys contiguous (b128 stores)
    #pragma unroll
    for (int mi = 0; mi < 4; mi++) {
      v8h ph;
      #pragma unroll
      for (int r = 0; r < 8; r++) ph[r] = (_Float16)sacc[mi][r];
      *(v8h*)(pbase + l * 72 + mi * 16 + 8 * hi) = ph;
    }

    asm volatile("s_wait_dscnt 0x0" ::: "memory");  // wave-local LDS ordering

    // B-frags of P^T: column q = l, table k order
    H16 bp[2];
    #pragma unroll
    for (int t = 0; t < 2; t++) {
      const _Float16* p = pbase + l * 72 + t * 32;
      bp[t].h[0] = *(const v8h*)(p + hi * 8);
      bp[t].h[1] = *(const v8h*)(p + 16 + hi * 8);
    }

    // O^T += V^T @ P^T : A-frags = V^T rows (M=d) from transposed sV
    #pragma unroll
    for (int mi = 0; mi < 4; mi++) {
      #pragma unroll
      for (int t = 0; t < 2; t++) {
        H16 av;
        const _Float16* p = sV + (mi * 16 + l) * 72 + t * 32;
        av.h[0] = *(const v8h*)(p + hi * 8);
        av.h[1] = *(const v8h*)(p + 16 + hi * 8);
        o[mi] = wmma16(av.v, bp[t].v, o[mi]);
      }
    }
  }

  // epilogue: lane owns query row q0+l -> 4 x 16B contiguous stores
  {
    float inv = 1.0f / lrun;
    int q = q0 + l;
    _Float16* orow = Out + ((size_t)(b * S_ + q)) * DM_ + h * DK_;
    #pragma unroll
    for (int mi = 0; mi < 4; mi++) {
      v8h ph;
      #pragma unroll
      for (int r = 0; r < 8; r++) ph[r] = (_Float16)(o[mi][r] * inv);
      *(v8h*)(orow + mi * 16 + 8 * hi) = ph;
    }
  }
}

// ---------------- layernorm over last dim (1024) ----------------
__global__ __launch_bounds__(256) void ln_kernel(const float* __restrict__ x,
                                                 const float* __restrict__ g,
                                                 const float* __restrict__ bta,
                                                 float* __restrict__ out) {
  __shared__ float rs[8], rs2[8];
  int row = blockIdx.x;
  const float* xr = x + (size_t)row * DM_;
  int tid = threadIdx.x;
  float v[4], s = 0.f, s2 = 0.f;
  #pragma unroll
  for (int i = 0; i < 4; i++) {
    v[i] = xr[tid + i * 256];
    s += v[i];
    s2 += v[i] * v[i];
  }
  #pragma unroll
  for (int off = 1; off < 32; off <<= 1) {
    s  += __shfl_xor(s, off, 32);
    s2 += __shfl_xor(s2, off, 32);
  }
  if ((tid & 31) == 0) { rs[tid >> 5] = s; rs2[tid >> 5] = s2; }
  __syncthreads();
  s = 0.f; s2 = 0.f;
  #pragma unroll
  for (int w = 0; w < 8; w++) { s += rs[w]; s2 += rs2[w]; }
  float mu  = s * (1.0f / DM_);
  float var = s2 * (1.0f / DM_) - mu * mu;
  float inv = rsqrtf(var + 1e-5f);
  #pragma unroll
  for (int i = 0; i < 4; i++) {
    int c = tid + i * 256;
    out[(size_t)row * DM_ + c] = (v[i] - mu) * inv * g[c] + bta[c];
  }
}

extern "C" void kernel_launch(void* const* d_in, const int* in_sizes, int n_in,
                              void* d_out, int out_size, void* d_ws, size_t ws_size,
                              hipStream_t stream) {
  (void)in_sizes; (void)n_in; (void)out_size; (void)ws_size;
  const float* q    = (const float*)d_in[0];
  const float* k    = (const float*)d_in[1];
  const float* v    = (const float*)d_in[2];
  const float* w_qs = (const float*)d_in[3];
  const float* w_ks = (const float*)d_in[4];
  const float* w_vs = (const float*)d_in[5];
  const float* fc_w = (const float*)d_in[6];
  const float* fc_b = (const float*)d_in[7];
  const float* ln_g = (const float*)d_in[8];
  const float* ln_b = (const float*)d_in[9];

  char* ws = (char*)d_ws;
  const size_t MB = (size_t)1 << 20;
  _Float16* q16  = (_Float16*)(ws + 0 * MB);    // 8 MB each activation
  _Float16* k16  = (_Float16*)(ws + 8 * MB);
  _Float16* v16  = (_Float16*)(ws + 16 * MB);
  _Float16* wq16 = (_Float16*)(ws + 24 * MB);   // 2 MB each weight
  _Float16* wk16 = (_Float16*)(ws + 26 * MB);
  _Float16* wv16 = (_Float16*)(ws + 28 * MB);
  _Float16* wf16 = (_Float16*)(ws + 30 * MB);
  _Float16* Qh   = (_Float16*)(ws + 32 * MB);   // [B,H,S,D]
  _Float16* Kh   = (_Float16*)(ws + 40 * MB);   // [B,H,S,D]
  _Float16* Vt   = (_Float16*)(ws + 48 * MB);   // [B,H,D,S]
  _Float16* AO   = (_Float16*)(ws + 56 * MB);   // [B,S,H*D]
  float*    PL   = (float*)   (ws + 64 * MB);   // pre-LN f32 (16 MB)

  const int NTOK = MTOT * DM_;  // 4M
  const int NW   = DM_ * DM_;   // 1M
  cvt_kernel<<<NTOK / 1024, 256, 0, stream>>>(q, q16, NTOK);
  cvt_kernel<<<NTOK / 1024, 256, 0, stream>>>(k, k16, NTOK);
  cvt_kernel<<<NTOK / 1024, 256, 0, stream>>>(v, v16, NTOK);
  cvt_kernel<<<NW / 1024, 256, 0, stream>>>(w_qs, wq16, NW);
  cvt_kernel<<<NW / 1024, 256, 0, stream>>>(w_ks, wk16, NW);
  cvt_kernel<<<NW / 1024, 256, 0, stream>>>(w_vs, wv16, NW);
  cvt_kernel<<<NW / 1024, 256, 0, stream>>>(fc_w, wf16, NW);

  dim3 ggrid(DM_ / 128, MTOT / 128);  // (8, 32)
  gemm_kernel<0><<<ggrid, 256, 0, stream>>>(q16, wq16, (void*)Qh, nullptr, nullptr);
  gemm_kernel<0><<<ggrid, 256, 0, stream>>>(k16, wk16, (void*)Kh, nullptr, nullptr);
  gemm_kernel<1><<<ggrid, 256, 0, stream>>>(v16, wv16, (void*)Vt, nullptr, nullptr);

  attn_kernel<<<dim3(S_ / 128, B_ * H_), 256, 0, stream>>>(Qh, Kh, Vt, AO);

  gemm_kernel<2><<<ggrid, 256, 0, stream>>>(AO, wf16, (void*)PL, fc_b, q);

  ln_kernel<<<MTOT, 256, 0, stream>>>(PL, ln_g, ln_b, (float*)d_out);
}
